// Attention_65206193488491
// MI455X (gfx1250) — matrix operand order repaired and checked by reading, not compiled, not run on hardware
//
#include <hip/hip_runtime.h>

// ---------------------------------------------------------------------------
// Transformer-XL relative attention block for gfx1250 (MI455X), wave32 WMMA.
// B=2, Tq=1024, Tkv=2048, H=16, Dv=64, Dmodel=1024.
// ---------------------------------------------------------------------------

#define BB    2
#define TQL   1024
#define TKVL  2048
#define NH    16
#define DM    1024
#define DV    64
#define ROFF  (TKVL - TQL)   // 1024

typedef _Float16 v16h __attribute__((ext_vector_type(16)));
typedef _Float16 v8h  __attribute__((ext_vector_type(8)));
typedef _Float16 h4   __attribute__((ext_vector_type(4)));
typedef float    v8f  __attribute__((ext_vector_type(8)));

static __device__ __forceinline__ v16h concat8(v8h lo, v8h hi) {
  return __builtin_shufflevector(lo, hi, 0,1,2,3,4,5,6,7,8,9,10,11,12,13,14,15);
}

static __device__ __forceinline__ v8f wmma_f16(v16h a, v16h b, v8f c) {
  // D = A(16x32 f16) x B(32x16 f16) + C(16x16 f32)
  return __builtin_amdgcn_wmma_f32_16x16x32_f16(false, a, false, b, (short)0, c,
                                                false, false);
}

// ---------------------------------------------------------------------------
// Kernel 1: projection GEMM   Y = X @ W   -> head-major f16 [B][H][T][DV]
// Templated on log2(T) (division-free epilogue) and dual-output (q+cb, q+pb).
// Tile: 64(M) x 64(N), K-step 32 (unroll 2), 128 threads / 4 waves.
// ---------------------------------------------------------------------------
template <int TLOG2, bool DUAL>
__global__ __launch_bounds__(128) void proj_kernel(
    const float* __restrict__ X,   // [B*T, DM]
    const float* __restrict__ W,   // [DM, DM]
    _Float16* __restrict__ out1, const float* __restrict__ bias1,
    _Float16* __restrict__ out2, const float* __restrict__ bias2)
{
  constexpr int T = 1 << TLOG2;
  __shared__ _Float16 As[64 * 32];   // [row][k]
  __shared__ _Float16 Bs[64 * 32];   // [col][k]  (transposed W tile)

  const int tid  = threadIdx.x;
  const int lane = tid & 31;
  const int wave = tid >> 5;
  const int l15  = lane & 15;
  const int hi   = lane >> 4;

  const int m0 = blockIdx.x * 64;
  const int h  = blockIdx.y;        // 64-wide N tile == one head
  const int n0 = h * DV;

  const int wr = (wave >> 1) * 32;
  const int wc = (wave & 1) * 32;

  const int bcol  = tid & 63;
  const int bhalf = tid >> 6;

  v8f acc[2][2];
#pragma unroll
  for (int i = 0; i < 2; ++i)
#pragma unroll
    for (int j = 0; j < 2; ++j) acc[i][j] = {};

#pragma unroll 2
  for (int k0 = 0; k0 < DM; k0 += 32) {
    // ---- stage A tile (64x32 fp32 -> f16) ----
#pragma unroll
    for (int it = 0; it < 4; ++it) {
      int f4  = tid + 128 * it;          // 0..511 float4s
      int row = f4 >> 3;
      int c4  = (f4 & 7) * 4;
      const float* gp = X + (size_t)(m0 + row) * DM + k0 + c4;
      float4 v = *(const float4*)gp;
      __builtin_prefetch(gp + 32, 0, 0);   // next K tile -> global_prefetch_b8
      h4 hv;
      hv.x = (_Float16)v.x; hv.y = (_Float16)v.y;
      hv.z = (_Float16)v.z; hv.w = (_Float16)v.w;
      *(h4*)&As[row * 32 + c4] = hv;
    }
    // ---- stage B tile transposed (32x64 -> [col][k]) ----
#pragma unroll
    for (int i = 0; i < 16; ++i) {
      int k = bhalf * 16 + i;
      const float* gp = W + (size_t)(k0 + k) * DM + n0 + bcol;
      __builtin_prefetch(gp + 32 * DM, 0, 0);
      Bs[bcol * 32 + k] = (_Float16)(*gp);
    }
    __syncthreads();

    v16h af[2], bf[2];
#pragma unroll
    for (int rt = 0; rt < 2; ++rt) {
      const _Float16* ap = &As[(wr + rt * 16 + l15) * 32 + hi * 8];
      af[rt] = concat8(*(const v8h*)ap, *(const v8h*)(ap + 16));
    }
#pragma unroll
    for (int ct = 0; ct < 2; ++ct) {
      const _Float16* bp = &Bs[(wc + ct * 16 + l15) * 32 + hi * 8];
      bf[ct] = concat8(*(const v8h*)bp, *(const v8h*)(bp + 16));
    }
#pragma unroll
    for (int rt = 0; rt < 2; ++rt)
#pragma unroll
      for (int ct = 0; ct < 2; ++ct)
        acc[rt][ct] = wmma_f16(af[rt], bf[ct], acc[rt][ct]);
    __syncthreads();
  }

  // ---- store head-major f16 (+ per-(h,dv) biases) ----
#pragma unroll
  for (int rt = 0; rt < 2; ++rt)
#pragma unroll
    for (int ct = 0; ct < 2; ++ct) {
      int dv = wc + ct * 16 + l15;
      float b1 = DUAL ? bias1[h * DV + dv] : 0.f;
      float b2 = DUAL ? bias2[h * DV + dv] : 0.f;
#pragma unroll
      for (int r = 0; r < 8; ++r) {
        int m = m0 + wr + rt * 16 + hi * 8 + r;
        int b = m >> TLOG2, t = m & (T - 1);
        size_t idx = (((size_t)b * NH + h) * T + t) * DV + dv;
        float v = acc[rt][ct][r];
        out1[idx] = (_Float16)(v + b1);
        if (DUAL) out2[idx] = (_Float16)(v + b2);
      }
    }
}

// ---------------------------------------------------------------------------
// Kernel 2: relative attention with flash-style online softmax.
// Grid: (TQ/64, NH, B). Block: 128 threads = 4 waves; each wave owns 16 q-rows.
// Per 64-col k-tile: ac = (q+cb)K^T via WMMA, S strip = (q+pb)R^T over the
// 128-wide skewed band via WMMA, diagonal gather for rel_shift, mask, online
// softmax, then probs x V via WMMA into the ctx accumulator.
// ---------------------------------------------------------------------------
__global__ __launch_bounds__(128) void attn_kernel(
    const _Float16* __restrict__ Qc, const _Float16* __restrict__ Qp,
    const _Float16* __restrict__ Kh, const _Float16* __restrict__ Vh,
    const _Float16* __restrict__ Rh, float* __restrict__ Ctx)
{
  __shared__ _Float16 Ks[64 * 64];        // [j][dv]          8 KB
  __shared__ _Float16 Vt[64 * 64];        // [dv][j]          8 KB
  __shared__ _Float16 Rs[128 * 64];       // [p][dv]         16 KB
  __shared__ _Float16 Ssm[4][16 * 128];   // per-wave S strip 16 KB
  __shared__ _Float16 Ps[4][16 * 64];     // per-wave probs    8 KB

  const int tid  = threadIdx.x;
  const int lane = tid & 31;
  const int wave = tid >> 5;
  const int l15  = lane & 15;
  const int hi   = lane >> 4;

  const int b  = blockIdx.z;
  const int h  = blockIdx.y;
  const int i0 = blockIdx.x * 64;
  const size_t bh = (size_t)b * NH + h;

  // preload Q fragments (A-matrix, reused for all k-tiles)
  v16h aQc[2], aQp[2];
  {
    const _Float16* qc = Qc + (bh * TQL + i0 + wave * 16 + l15) * DV;
    const _Float16* qp = Qp + (bh * TQL + i0 + wave * 16 + l15) * DV;
#pragma unroll
    for (int ks = 0; ks < 2; ++ks) {
      const _Float16* p1 = qc + ks * 32 + hi * 8;
      aQc[ks] = concat8(*(const v8h*)p1, *(const v8h*)(p1 + 16));
      const _Float16* p2 = qp + ks * 32 + hi * 8;
      aQp[ks] = concat8(*(const v8h*)p2, *(const v8h*)(p2 + 16));
    }
  }

  v8f acc[4];
#pragma unroll
  for (int t = 0; t < 4; ++t) acc[t] = {};
  float m_run[8], l_run[8];
#pragma unroll
  for (int r = 0; r < 8; ++r) { m_run[r] = -1e30f; l_run[r] = 0.f; }

  const int jt_max = 17 + blockIdx.x;   // causal truncation of k-tiles

  for (int jt = 0; jt < jt_max; ++jt) {
    const int j0 = jt * 64;

    // ---- cooperative stage: K tile, transposed V tile ----
    {
      int row = tid >> 1, part = tid & 1;
      const _Float16* kp = Kh + (bh * TKVL + j0 + row) * DV + part * 32;
      const _Float16* vp = Vh + (bh * TKVL + j0 + row) * DV + part * 32;
      __builtin_prefetch(kp + 64 * DV, 0, 0);   // next j-tile
      __builtin_prefetch(vp + 64 * DV, 0, 0);
      v8h kv[4], vv[4];
#pragma unroll
      for (int i = 0; i < 4; ++i) { kv[i] = *(const v8h*)(kp + i * 8);
                                    vv[i] = *(const v8h*)(vp + i * 8); }
#pragma unroll
      for (int i = 0; i < 4; ++i) *(v8h*)&Ks[row * 64 + part * 32 + i * 8] = kv[i];
#pragma unroll
      for (int i = 0; i < 4; ++i)
#pragma unroll
        for (int e = 0; e < 8; ++e)
          Vt[(part * 32 + i * 8 + e) * 64 + row] = vv[i][e];
    }
    // ---- stage R strip: rows p = pmin..pmin+127, pmin = j0-i0+960 ----
    {
      int p = j0 - i0 + 960 + tid;
      p = p < 0 ? 0 : (p > TKVL - 1 ? TKVL - 1 : p);
      const _Float16* rp = Rh + (bh * TKVL + p) * DV;
      __builtin_prefetch(rp + 64 * DV, 0, 0);
#pragma unroll
      for (int i = 0; i < 8; ++i)
        *(v8h*)&Rs[tid * 64 + i * 8] = *(const v8h*)(rp + i * 8);
    }
    __syncthreads();

    // ---- ac = (q+cb) . k  (4 col tiles x 2 K-steps) ----
    v8f sc[4];
#pragma unroll
    for (int ct = 0; ct < 4; ++ct) {
      v8f s = {};
#pragma unroll
      for (int ks = 0; ks < 2; ++ks) {
        const _Float16* bp = &Ks[(ct * 16 + l15) * 64 + ks * 32 + hi * 8];
        v16h bf = concat8(*(const v8h*)bp, *(const v8h*)(bp + 16));
        s = wmma_f16(aQc[ks], bf, s);
      }
      sc[ct] = s;
    }

    // ---- S strip = (q+pb) . r over 128 band columns ----
#pragma unroll
    for (int pt = 0; pt < 8; ++pt) {
      v8f s = {};
#pragma unroll
      for (int ks = 0; ks < 2; ++ks) {
        const _Float16* bp = &Rs[(pt * 16 + l15) * 64 + ks * 32 + hi * 8];
        v16h bf = concat8(*(const v8h*)bp, *(const v8h*)(bp + 16));
        s = wmma_f16(aQp[ks], bf, s);
      }
#pragma unroll
      for (int r = 0; r < 8; ++r)
        Ssm[wave][(hi * 8 + r) * 128 + pt * 16 + l15] = (_Float16)s[r];
    }

    // ---- assemble logits: ac + rel-shifted bd, causal mask, tile max ----
    float tmax[8];
#pragma unroll
    for (int r = 0; r < 8; ++r) tmax[r] = -1e30f;
#pragma unroll
    for (int ct = 0; ct < 4; ++ct) {
      int jr = ct * 16 + l15;
#pragma unroll
      for (int r = 0; r < 8; ++r) {
        int m    = hi * 8 + r;
        int irow = i0 + wave * 16 + m;
        int sidx = jr - (wave * 16 + m) + 63;      // in [0,126]
        float lg = (sc[ct][r] + (float)Ssm[wave][m * 128 + sidx]) * 0.125f;
        if (j0 + jr > ROFF + irow) lg = -1e30f;    // causal mask
        sc[ct][r] = lg;
        tmax[r] = fmaxf(tmax[r], lg);
      }
    }
#pragma unroll
    for (int off = 8; off; off >>= 1)
#pragma unroll
      for (int r = 0; r < 8; ++r)
        tmax[r] = fmaxf(tmax[r], __shfl_xor(tmax[r], off));

    // ---- online softmax update ----
    float psum[8];
#pragma unroll
    for (int r = 0; r < 8; ++r) {
      float mnew = fmaxf(m_run[r], tmax[r]);
      float corr = __expf(m_run[r] - mnew);
      m_run[r] = mnew;
      l_run[r] *= corr;
      psum[r] = 0.f;
#pragma unroll
      for (int t = 0; t < 4; ++t) acc[t][r] *= corr;
    }
#pragma unroll
    for (int ct = 0; ct < 4; ++ct)
#pragma unroll
      for (int r = 0; r < 8; ++r) {
        float lg = sc[ct][r];
        float p  = (lg <= -1e29f) ? 0.f : __expf(lg - m_run[r]);
        psum[r] += p;
        Ps[wave][(hi * 8 + r) * 64 + ct * 16 + l15] = (_Float16)p;
      }
#pragma unroll
    for (int off = 8; off; off >>= 1)
#pragma unroll
      for (int r = 0; r < 8; ++r) psum[r] += __shfl_xor(psum[r], off);
#pragma unroll
    for (int r = 0; r < 8; ++r) l_run[r] += psum[r];

    // ---- ctx += P x V  (A from Ps, B from transposed V tile) ----
#pragma unroll
    for (int ks = 0; ks < 2; ++ks) {
      const _Float16* pp = &Ps[wave][l15 * 64 + ks * 32 + hi * 8];
      v16h pf = concat8(*(const v8h*)pp, *(const v8h*)(pp + 16));
#pragma unroll
      for (int dt = 0; dt < 4; ++dt) {
        const _Float16* vp = &Vt[(dt * 16 + l15) * 64 + ks * 32 + hi * 8];
        v16h vf = concat8(*(const v8h*)vp, *(const v8h*)(vp + 16));
        acc[dt] = wmma_f16(pf, vf, acc[dt]);
      }
    }
    __syncthreads();
  }

  // ---- normalize + write ctx [B][Tq][H*DV] fp32 ----
#pragma unroll
  for (int r = 0; r < 8; ++r) {
    float inv = 1.0f / l_run[r];
    int irow = i0 + wave * 16 + hi * 8 + r;
    float* op = Ctx + ((size_t)b * TQL + irow) * DM + h * DV;
#pragma unroll
    for (int dt = 0; dt < 4; ++dt)
      op[dt * 16 + l15] = acc[dt][r] * inv;
  }
}

// ---------------------------------------------------------------------------
// Kernel 3: out = ctx @ Wo + residual  (fp32 out into d_out, pre-LayerNorm)
// ---------------------------------------------------------------------------
__global__ __launch_bounds__(128) void out_gemm_kernel(
    const float* __restrict__ Ctx, const float* __restrict__ Wo,
    const float* __restrict__ resid, float* __restrict__ out)
{
  __shared__ _Float16 As[64 * 32];
  __shared__ _Float16 Bs[64 * 32];

  const int tid  = threadIdx.x;
  const int lane = tid & 31;
  const int wave = tid >> 5;
  const int l15  = lane & 15;
  const int hi   = lane >> 4;

  const int m0 = blockIdx.x * 64;
  const int n0 = blockIdx.y * 64;
  const int wr = (wave >> 1) * 32;
  const int wc = (wave & 1) * 32;
  const int bcol  = tid & 63;
  const int bhalf = tid >> 6;

  v8f acc[2][2];
#pragma unroll
  for (int i = 0; i < 2; ++i)
#pragma unroll
    for (int j = 0; j < 2; ++j) acc[i][j] = {};

#pragma unroll 2
  for (int k0 = 0; k0 < DM; k0 += 32) {
#pragma unroll
    for (int it = 0; it < 4; ++it) {
      int f4 = tid + 128 * it;
      int row = f4 >> 3;
      int c4 = (f4 & 7) * 4;
      const float* gp = Ctx + (size_t)(m0 + row) * DM + k0 + c4;
      float4 v = *(const float4*)gp;
      __builtin_prefetch(gp + 32, 0, 0);
      h4 hv;
      hv.x = (_Float16)v.x; hv.y = (_Float16)v.y;
      hv.z = (_Float16)v.z; hv.w = (_Float16)v.w;
      *(h4*)&As[row * 32 + c4] = hv;
    }
#pragma unroll
    for (int i = 0; i < 16; ++i) {
      int k = bhalf * 16 + i;
      const float* gp = Wo + (size_t)(k0 + k) * DM + n0 + bcol;
      __builtin_prefetch(gp + 32 * DM, 0, 0);
      Bs[bcol * 32 + k] = (_Float16)(*gp);
    }
    __syncthreads();

    v16h af[2], bf[2];
#pragma unroll
    for (int rt = 0; rt < 2; ++rt) {
      const _Float16* ap = &As[(wr + rt * 16 + l15) * 32 + hi * 8];
      af[rt] = concat8(*(const v8h*)ap, *(const v8h*)(ap + 16));
    }
#pragma unroll
    for (int ct = 0; ct < 2; ++ct) {
      const _Float16* bp = &Bs[(wc + ct * 16 + l15) * 32 + hi * 8];
      bf[ct] = concat8(*(const v8h*)bp, *(const v8h*)(bp + 16));
    }
#pragma unroll
    for (int rt = 0; rt < 2; ++rt)
#pragma unroll
      for (int ct = 0; ct < 2; ++ct)
        acc[rt][ct] = wmma_f16(af[rt], bf[ct], acc[rt][ct]);
    __syncthreads();
  }

#pragma unroll
  for (int rt = 0; rt < 2; ++rt)
#pragma unroll
    for (int ct = 0; ct < 2; ++ct) {
      int n = n0 + wc + ct * 16 + l15;
#pragma unroll
      for (int r = 0; r < 8; ++r) {
        int m = m0 + wr + rt * 16 + hi * 8 + r;
        size_t idx = (size_t)m * DM + n;
        out[idx] = acc[rt][ct][r] + resid[idx];
      }
    }
}

// ---------------------------------------------------------------------------
// Kernel 4: in-place LayerNorm over the last dim (1024), one block per row.
// ---------------------------------------------------------------------------
__global__ __launch_bounds__(256) void ln_kernel(
    float* __restrict__ out, const float* __restrict__ gamma,
    const float* __restrict__ beta)
{
  __shared__ float red[16];
  float* p = out + (size_t)blockIdx.x * DM;

  float x[4], s = 0.f, s2 = 0.f;
#pragma unroll
  for (int i = 0; i < 4; ++i) {
    x[i] = p[threadIdx.x + i * 256];
    s += x[i]; s2 += x[i] * x[i];
  }
#pragma unroll
  for (int off = 16; off; off >>= 1) {
    s  += __shfl_xor(s, off);
    s2 += __shfl_xor(s2, off);
  }
  int wv = threadIdx.x >> 5, ln = threadIdx.x & 31;
  if (ln == 0) { red[wv] = s; red[8 + wv] = s2; }
  __syncthreads();
  float ts = 0.f, ts2 = 0.f;
#pragma unroll
  for (int w = 0; w < 8; ++w) { ts += red[w]; ts2 += red[8 + w]; }
  float mu  = ts / (float)DM;
  float var = ts2 / (float)DM - mu * mu;
  float rs  = rsqrtf(var + 1e-5f);
#pragma unroll
  for (int i = 0; i < 4; ++i) {
    int c = threadIdx.x + i * 256;
    p[c] = (x[i] - mu) * rs * gamma[c] + beta[c];
  }
}

// ---------------------------------------------------------------------------
// Host-side launcher
// ---------------------------------------------------------------------------
extern "C" void kernel_launch(void* const* d_in, const int* in_sizes, int n_in,
                              void* d_out, int out_size, void* d_ws, size_t ws_size,
                              hipStream_t stream) {
  (void)in_sizes; (void)n_in; (void)out_size; (void)ws_size;

  const float* query  = (const float*)d_in[0];   // [B,Tq,DM]
  const float* keyval = (const float*)d_in[1];   // [B,Tkv,DM]
  const float* relpos = (const float*)d_in[2];   // [B,Tkv,DM]
  const float* cb     = (const float*)d_in[3];   // [H,DV]
  const float* pb     = (const float*)d_in[4];   // [H,DV]
  const float* Wq     = (const float*)d_in[5];
  const float* Wk     = (const float*)d_in[6];
  const float* Wv     = (const float*)d_in[7];
  const float* Wr     = (const float*)d_in[8];
  const float* Wo     = (const float*)d_in[9];
  const float* gamma  = (const float*)d_in[10];
  const float* beta   = (const float*)d_in[11];
  // d_in[12] (mask) is recomputed analytically: j > ROFF + i.
  float* out = (float*)d_out;

  // workspace layout (40 MiB total)
  char* ws = (char*)d_ws;
  _Float16* Qc = (_Float16*)ws;                    ws += (size_t)BB * NH * TQL  * DV * 2;
  _Float16* Qp = (_Float16*)ws;                    ws += (size_t)BB * NH * TQL  * DV * 2;
  _Float16* Kh = (_Float16*)ws;                    ws += (size_t)BB * NH * TKVL * DV * 2;
  _Float16* Vh = (_Float16*)ws;                    ws += (size_t)BB * NH * TKVL * DV * 2;
  _Float16* Rh = (_Float16*)ws;                    ws += (size_t)BB * NH * TKVL * DV * 2;
  float*    Ctx = (float*)ws;

  dim3 blk(128);
  // projections (q gets both content- and position-biased copies)
  proj_kernel<10, true><<<dim3(BB * TQL / 64, NH), blk, 0, stream>>>(
      query, Wq, Qc, cb, Qp, pb);
  proj_kernel<11, false><<<dim3(BB * TKVL / 64, NH), blk, 0, stream>>>(
      keyval, Wk, Kh, nullptr, nullptr, nullptr);
  proj_kernel<11, false><<<dim3(BB * TKVL / 64, NH), blk, 0, stream>>>(
      keyval, Wv, Vh, nullptr, nullptr, nullptr);
  proj_kernel<11, false><<<dim3(BB * TKVL / 64, NH), blk, 0, stream>>>(
      relpos, Wr, Rh, nullptr, nullptr, nullptr);

  // attention
  attn_kernel<<<dim3(TQL / 64, NH, BB), blk, 0, stream>>>(Qc, Qp, Kh, Vh, Rh, Ctx);

  // output projection + residual
  out_gemm_kernel<<<dim3(BB * TQL / 64, DM / 64), blk, 0, stream>>>(Ctx, Wo, query, out);

  // layernorm in-place
  ln_kernel<<<dim3(BB * TQL), dim3(256), 0, stream>>>(out, gamma, beta);
}
